// BatchedBottleneckAttentionProbes_33852932227691
// MI455X (gfx1250) — matrix operand order repaired and checked by reading, not compiled, not run on hardware
//
#include <hip/hip_runtime.h>
#include <hip/hip_bf16.h>

typedef _Float16 v16h __attribute__((ext_vector_type(16)));
typedef __fp16   h2v  __attribute__((ext_vector_type(2)));
typedef float    v8f  __attribute__((ext_vector_type(8)));
typedef float    v4f  __attribute__((ext_vector_type(4)));

namespace {
constexpr int NB  = 16;    // batch
constexpr int NS  = 4096;  // sequence length (== MAX_S)
constexpr int ND  = 512;   // model dim
constexpr int NP  = 64;    // probes
constexpr int NBD = 32;    // bottleneck dim
constexpr int NH  = 4;     // heads
constexpr int NHD = 8;     // head dim
constexpr float LN_EPS = 1e-5f;

union H2U { h2v h; unsigned int u; };
union AU  { v16h v; h2v h[8]; };
} // namespace

__device__ __forceinline__ float gelu_exact(float x) {
  // torch F.gelu default (erf form)
  return 0.5f * x * (1.0f + erff(x * 0.70710678118654752440f));
}

__device__ __forceinline__ float hsum16(float v) {
  // butterfly sum within each 16-lane half of the wave32
  v += __shfl_xor(v, 1, 32);
  v += __shfl_xor(v, 2, 32);
  v += __shfl_xor(v, 4, 32);
  v += __shfl_xor(v, 8, 32);
  return v;
}

__launch_bounds__(256)
__global__ void bottleneck_probes_wmma_kernel(
    const float* __restrict__ x,          // [B,S,D]
    const unsigned char* __restrict__ mask, // [B,S] (bool)
    const float* __restrict__ proj_w,     // [P,BD,D]
    const float* __restrict__ proj_b,     // [P,BD]
    const float* __restrict__ ln_w,       // [P,BD]
    const float* __restrict__ ln_b,       // [P,BD]
    const float* __restrict__ query_w,    // [P,H,BD]
    const float* __restrict__ query_b,    // [P,H]
    const float* __restrict__ pos_bias,   // [P,H,MAX_S]
    const float* __restrict__ value_w,    // [P,H,HD,BD]
    const float* __restrict__ value_b,    // [P,H,HD]
    const float* __restrict__ class_w,    // [P,1,H*HD]
    const float* __restrict__ class_b,    // [P,1]
    float* __restrict__ out)              // [B,P]
{
  // ---------------- LDS ----------------
  // B-operand: proj_w[p] as f16 in this kernel's B layout (contiguous K per half):
  //   Blds[((t*2 + j)*32 + lane)*8 + v]  (t = k-step 0..15, j = N-tile 0..1)
  //   lane<16: N=lane,    K = 32t + {0..15}   (dword v = pair K=2v,2v+1)
  //   lane>=16:N=lane-16, K = 32t + 16 + {0..15}
  __shared__ unsigned int Blds[16 * 2 * 32 * 8];       // 32 KB
  __shared__ float mW[16][NH];                          // per (wave,half) group
  __shared__ float lW[16][NH];
  __shared__ float accW[16][NH][NBD];                   // 8 KB
  __shared__ float waccS[NH][NBD];
  __shared__ float partS[NH * NHD];

  const int p    = blockIdx.x & (NP - 1);
  const int b    = blockIdx.x >> 6;
  const int tid  = threadIdx.x;
  const int lane = tid & 31;
  const int w    = tid >> 5;       // wave id 0..7
  const int hf   = lane >> 4;      // 16-lane half
  const int lx   = lane & 15;

  // ---- Phase 0: convert proj_w[p] (f32 row-major) -> LDS f16 B layout ----
  {
    const float* Wp = proj_w + (size_t)p * NBD * ND;
    for (int i = tid; i < NBD * ND / 2; i += 256) {
      int n  = i >> 8;            // 256 f16-pairs per bottleneck row
      int pr = i & 255;
      int d0 = pr << 1;           // even K
      int t  = d0 >> 5;
      int r  = d0 & 31;
      int kh = r >> 4;            // K half
      int v  = (r & 15) >> 1;
      int j  = n >> 4;
      int bl = (n & 15) + (kh << 4);
      H2U pk;
      pk.h = __builtin_amdgcn_cvt_pkrtz(Wp[n * ND + d0], Wp[n * ND + d0 + 1]);
      Blds[(((t * 2 + j) * 32 + bl) << 3) + v] = pk.u;
    }
  }

  // per-lane constants for the two N columns this lane owns in C layout
  const int N0 = lx;
  const int N1 = 16 + lx;
  const float pbc0 = proj_b[p * NBD + N0];
  const float pbc1 = proj_b[p * NBD + N1];
  const float lw0  = ln_w[p * NBD + N0];
  const float lw1  = ln_w[p * NBD + N1];
  const float lb0  = ln_b[p * NBD + N0];
  const float lb1  = ln_b[p * NBD + N1];
  float qw0[NH], qw1[NH], qb[NH];
#pragma unroll
  for (int hh = 0; hh < NH; ++hh) {
    qw0[hh] = query_w[(p * NH + hh) * NBD + N0];
    qw1[hh] = query_w[(p * NH + hh) * NBD + N1];
    qb[hh]  = query_b[p * NH + hh];
  }

  __syncthreads();

  // ---- Phase 1: fused GEMM + LN + gelu + online softmax over S ----
  const float NEG_INF = -__builtin_inff();
  float m[NH], l[NH], a0[NH], a1[NH];
#pragma unroll
  for (int hh = 0; hh < NH; ++hh) { m[hh] = NEG_INF; l[hh] = 0.f; a0[hh] = 0.f; a1[hh] = 0.f; }

  const float* xb = x + (size_t)b * NS * ND;

  for (int mt = w; mt < NS / 16; mt += 8) {
    const int s0  = mt * 16;
    const int row = s0 + lx;                 // A-layout: M = lane&15 for both halves
    const float* xr = xb + (size_t)row * ND;
    // prefetch this lane's row for the next iteration (mt+8 -> row+128)
    __builtin_prefetch(xr + 128 * ND, 0, 1);

    v8f c0 = {0.f, 0.f, 0.f, 0.f, 0.f, 0.f, 0.f, 0.f};
    v8f c1 = {0.f, 0.f, 0.f, 0.f, 0.f, 0.f, 0.f, 0.f};

#pragma unroll
    for (int t = 0; t < 16; ++t) {
      const int kb = t * 32 + hf * 16;       // A-layout K base for this half
      v4f A0 = *(const v4f*)(xr + kb);
      v4f A1 = *(const v4f*)(xr + kb + 4);
      v4f A2 = *(const v4f*)(xr + kb + 8);
      v4f A3 = *(const v4f*)(xr + kb + 12);
      AU a;
      a.h[0] = __builtin_amdgcn_cvt_pkrtz(A0.x, A0.y);
      a.h[1] = __builtin_amdgcn_cvt_pkrtz(A0.z, A0.w);
      a.h[2] = __builtin_amdgcn_cvt_pkrtz(A1.x, A1.y);
      a.h[3] = __builtin_amdgcn_cvt_pkrtz(A1.z, A1.w);
      a.h[4] = __builtin_amdgcn_cvt_pkrtz(A2.x, A2.y);
      a.h[5] = __builtin_amdgcn_cvt_pkrtz(A2.z, A2.w);
      a.h[6] = __builtin_amdgcn_cvt_pkrtz(A3.x, A3.y);
      a.h[7] = __builtin_amdgcn_cvt_pkrtz(A3.z, A3.w);

      const v16h b0 = *(const v16h*)&Blds[((t * 2 + 0) * 32 + lane) << 3];
      const v16h b1 = *(const v16h*)&Blds[((t * 2 + 1) * 32 + lane) << 3];

      c0 = __builtin_amdgcn_wmma_f32_16x16x32_f16(false, a.v, false, b0,
                                                  (short)0, c0, false, false);
      c1 = __builtin_amdgcn_wmma_f32_16x16x32_f16(false, a.v, false, b1,
                                                  (short)0, c1, false, false);
    }

    // C layout: VGPR r, this half -> row s0 + hf*8 + r ; columns N0 (c0) / N1 (c1)
    // LayerNorm over BD=32 (split across 16 lanes x 2 regs) + exact gelu
    float g0[8], g1[8];
#pragma unroll
    for (int r = 0; r < 8; ++r) {
      float h0 = c0[r] + pbc0;
      float h1 = c1[r] + pbc1;
      float mu = hsum16(h0 + h1) * (1.0f / 32.0f);
      float d0v = h0 - mu, d1v = h1 - mu;
      float var = hsum16(d0v * d0v + d1v * d1v) * (1.0f / 32.0f);
      float rstd = rsqrtf(var + LN_EPS);
      g0[r] = gelu_exact(d0v * rstd * lw0 + lb0);
      g1[r] = gelu_exact(d1v * rstd * lw1 + lb1);
    }

    // mask bytes for this half's 8 rows
    const unsigned char* mrow = mask + b * NS + s0 + hf * 8;
    const unsigned int mk0 = *(const unsigned int*)(mrow);
    const unsigned int mk1 = *(const unsigned int*)(mrow + 4);

    // per-head logits + flash-style online softmax update (state per half-wave)
#pragma unroll
    for (int hh = 0; hh < NH; ++hh) {
      const float* pbase = pos_bias + ((size_t)(p * NH + hh)) * NS + s0 + hf * 8;
      v4f pA = *(const v4f*)pbase;
      v4f pB = *(const v4f*)(pbase + 4);
      float lt[8];
#pragma unroll
      for (int r = 0; r < 8; ++r)
        lt[r] = hsum16(g0[r] * qw0[hh] + g1[r] * qw1[hh]) + qb[hh];
      lt[0] += pA.x; lt[1] += pA.y; lt[2] += pA.z; lt[3] += pA.w;
      lt[4] += pB.x; lt[5] += pB.y; lt[6] += pB.z; lt[7] += pB.w;
#pragma unroll
      for (int r = 0; r < 8; ++r) {
        unsigned int mb = (r < 4) ? ((mk0 >> (8 * r)) & 0xffu)
                                  : ((mk1 >> (8 * (r - 4))) & 0xffu);
        if (mb == 0u) lt[r] = NEG_INF;
      }
      float tm = lt[0];
#pragma unroll
      for (int r = 1; r < 8; ++r) tm = fmaxf(tm, lt[r]);
      float mn = fmaxf(m[hh], tm);
      float coef = (m[hh] == NEG_INF) ? 0.0f : __expf(m[hh] - mn);
      float ls = 0.f, s0a = 0.f, s1a = 0.f;
#pragma unroll
      for (int r = 0; r < 8; ++r) {
        float e = (lt[r] == NEG_INF) ? 0.0f : __expf(lt[r] - mn);
        ls  += e;
        s0a += e * g0[r];
        s1a += e * g1[r];
      }
      l[hh]  = l[hh]  * coef + ls;
      a0[hh] = a0[hh] * coef + s0a;
      a1[hh] = a1[hh] * coef + s1a;
      m[hh]  = mn;
    }
  }

  // ---- Phase 2: merge the 16 (wave, half) accumulator groups ----
  {
    const int g = w * 2 + hf;
#pragma unroll
    for (int hh = 0; hh < NH; ++hh) {
      accW[g][hh][N0] = a0[hh];
      accW[g][hh][N1] = a1[hh];
    }
    if (lx == 0) {
#pragma unroll
      for (int hh = 0; hh < NH; ++hh) { mW[g][hh] = m[hh]; lW[g][hh] = l[hh]; }
    }
  }
  __syncthreads();

  if (tid < NH * NBD) {           // 128 threads: one per (head, N)
    const int hh = tid >> 5;
    const int N  = tid & 31;
    float mg = NEG_INF;
#pragma unroll
    for (int g = 0; g < 16; ++g) mg = fmaxf(mg, mW[g][hh]);
    float lg = 0.f, ac = 0.f;
#pragma unroll
    for (int g = 0; g < 16; ++g) {
      float c = (mW[g][hh] == NEG_INF) ? 0.0f : __expf(mW[g][hh] - mg);
      lg += lW[g][hh] * c;
      ac += accW[g][hh][N] * c;
    }
    waccS[hh][N] = ac / lg;       // weighted_h[b,p,hh,N]
  }
  __syncthreads();

  if (tid < NH * NHD) {           // 32 threads: one per (head, k)
    const int hh = tid >> 3;
    const int k  = tid & 7;
    const float* vw = value_w + (((size_t)(p * NH + hh)) * NHD + k) * NBD;
    float s = value_b[(p * NH + hh) * NHD + k];
#pragma unroll
    for (int N = 0; N < NBD; ++N) s += waccS[hh][N] * vw[N];
    partS[tid] = s * class_w[p * (NH * NHD) + (hh * NHD + k)];
  }
  __syncthreads();

  if (tid == 0) {
    float s = class_b[p];
#pragma unroll
    for (int i = 0; i < NH * NHD; ++i) s += partS[i];
    out[b * NP + p] = s;
  }
}

extern "C" void kernel_launch(void* const* d_in, const int* in_sizes, int n_in,
                              void* d_out, int out_size, void* d_ws, size_t ws_size,
                              hipStream_t stream) {
  (void)in_sizes; (void)n_in; (void)d_ws; (void)ws_size; (void)out_size;
  const float* x          = (const float*)d_in[0];
  const unsigned char* mk = (const unsigned char*)d_in[1];
  const float* proj_w     = (const float*)d_in[2];
  const float* proj_b     = (const float*)d_in[3];
  const float* ln_w       = (const float*)d_in[4];
  const float* ln_b       = (const float*)d_in[5];
  const float* query_w    = (const float*)d_in[6];
  const float* query_b    = (const float*)d_in[7];
  const float* pos_bias   = (const float*)d_in[8];
  const float* value_w    = (const float*)d_in[9];
  const float* value_b    = (const float*)d_in[10];
  const float* class_w    = (const float*)d_in[11];
  const float* class_b    = (const float*)d_in[12];
  float* out              = (float*)d_out;

  dim3 grid(NB * NP);   // 1024 blocks: p fastest so 64 consecutive blocks share x[b] in L2
  dim3 block(256);      // 8 wave32s
  bottleneck_probes_wmma_kernel<<<grid, block, 0, stream>>>(
      x, mk, proj_w, proj_b, ln_w, ln_b, query_w, query_b, pos_bias,
      value_w, value_b, class_w, class_b, out);
}
